// MultiHeadAttention_2499670966274
// MI455X (gfx1250) — compile-verified, matrix operand order checked by reading, not run on hardware
//
#include <hip/hip_runtime.h>

// ---------------------------------------------------------------------------
// MHA forward for MI455X (gfx1250, wave32, WMMA bf16 16x16x32, f32 accum)
// B=4 T=2048 C=1024 H=16 D=64
// Strategy: bf16 everywhere compute-heavy, f32 accumulation + f32 softmax.
// GEMMs are LDS-free (weights pre-transposed; everything L2-resident on 192MB L2).
// Attention double-buffers V tiles with global_load_async_to_lds_b128 (ASYNCcnt).
// ---------------------------------------------------------------------------

typedef __bf16 bf16_t;
typedef __attribute__((ext_vector_type(16))) __bf16 v16bf;
typedef __attribute__((ext_vector_type(8)))  __bf16 v8bf;
typedef __attribute__((ext_vector_type(8)))  float  v8f;

static constexpr int Bc = 4;
static constexpr int Tc = 2048;
static constexpr int Cc = 1024;
static constexpr int Hc = 16;
static constexpr int Dc = 64;
static constexpr int Mrows = Bc * Tc;   // 8192

#define WMMA_BF16(a, b, c) \
  __builtin_amdgcn_wmma_f32_16x16x32_bf16(false, (a), false, (b), (short)0, (c), false, false)

// 16-bit A/B fragment: lane<16 holds K {kb..kb+7, kb+16..kb+23} with kb=0,
// lane>=16 with kb=8. Caller passes p already offset by kb.
static __device__ __forceinline__ v16bf load_frag(const bf16_t* p) {
  v8bf lo = *(const v8bf*)(p);
  v8bf hi = *(const v8bf*)(p + 16);
  v16bf f;
#pragma unroll
  for (int i = 0; i < 8; ++i) { f[i] = lo[i]; f[8 + i] = hi[i]; }
  return f;
}

// ---------------------------------------------------------------------------
// fp32 -> bf16 (linear)
// ---------------------------------------------------------------------------
__global__ void cvt_f32_bf16(const float* __restrict__ in, bf16_t* __restrict__ out, int n) {
  int i = blockIdx.x * blockDim.x + threadIdx.x;
  int stride = gridDim.x * blockDim.x;
  for (; i < n; i += stride) out[i] = (bf16_t)in[i];
}

// ---------------------------------------------------------------------------
// fp32 [rows][cols] -> bf16 transposed [cols][rows], coalesced both sides.
// Block 256 thr handles a 32x32 tile via padded LDS.
// ---------------------------------------------------------------------------
__global__ void __launch_bounds__(256)
cvt_transpose_f32_bf16(const float* __restrict__ in, bf16_t* __restrict__ out,
                       int rows, int cols) {
  __shared__ float tile[32][33];
  const int bx = blockIdx.x * 32;           // col base
  const int by = blockIdx.y * 32;           // row base
  const int tx = threadIdx.x & 31;
  const int ty = threadIdx.x >> 5;          // 0..7
#pragma unroll
  for (int i = 0; i < 4; ++i)
    tile[ty + i * 8][tx] = in[(size_t)(by + ty + i * 8) * cols + bx + tx];
  __syncthreads();
#pragma unroll
  for (int i = 0; i < 4; ++i)
    out[(size_t)(bx + ty + i * 8) * rows + by + tx] = (bf16_t)tile[tx][ty + i * 8];
}

// ---------------------------------------------------------------------------
// LDS-free WMMA GEMM:  C[M,N] = A[M,K] @ Bt[N,K]^T   (+ bias for f32 out)
// Block 256 thr = 8 waves (4 along M x 2 along N); wave tile 32x64;
// block tile 128x128; K-step 32; operands stream from L2.
// ---------------------------------------------------------------------------
template <bool BF16OUT>
__global__ void __launch_bounds__(256)
gemm_abt(const bf16_t* __restrict__ A,   // [Mdim][Kdim] row-major
         const bf16_t* __restrict__ Bt,  // [Ndim][Kdim] row-major (= B transposed)
         bf16_t* __restrict__ Cb, float* __restrict__ Cf,
         const float* __restrict__ bias, int Mdim, int Ndim, int Kdim) {
  const int wave = threadIdx.x >> 5;
  const int lane = threadIdx.x & 31;
  const int half = lane >> 4;      // 0 | 1
  const int l16  = lane & 15;
  const int kb   = half * 8;       // fragment K base per half-wave
  const int wm = wave & 3;         // 0..3  (M)
  const int wn = wave >> 2;        // 0..1  (N)

  const int rowA0 = blockIdx.y * 128 + wm * 32;
  const int colB0 = blockIdx.x * 128 + wn * 64;

  const bf16_t* a0p = A + (size_t)(rowA0 + l16) * Kdim + kb;
  const bf16_t* a1p = A + (size_t)(rowA0 + 16 + l16) * Kdim + kb;
  const bf16_t* bp[4];
#pragma unroll
  for (int j = 0; j < 4; ++j)
    bp[j] = Bt + (size_t)(colB0 + j * 16 + l16) * Kdim + kb;

  v8f acc[2][4] = {};

  for (int k0 = 0; k0 < Kdim; k0 += 32) {
    // pull next k-slab toward L0/L2 (global_prefetch_b8)
    __builtin_prefetch(a0p + k0 + 256, 0, 0);
    __builtin_prefetch(bp[0] + k0 + 256, 0, 0);

    v16bf a0 = load_frag(a0p + k0);
    v16bf a1 = load_frag(a1p + k0);
#pragma unroll
    for (int j = 0; j < 4; ++j) {
      v16bf bf = load_frag(bp[j] + k0);
      acc[0][j] = WMMA_BF16(a0, bf, acc[0][j]);
      acc[1][j] = WMMA_BF16(a1, bf, acc[1][j]);
    }
  }

  // C/D layout: VGPR r -> row (half ? 8+r : r), lane l16 -> col.
#pragma unroll
  for (int j = 0; j < 4; ++j) {
    const int col = colB0 + j * 16 + l16;
    const float bv = BF16OUT ? 0.0f : bias[col];
#pragma unroll
    for (int sub = 0; sub < 2; ++sub)
#pragma unroll
      for (int r = 0; r < 8; ++r) {
        const int row = rowA0 + sub * 16 + half * 8 + r;
        if (BF16OUT)
          Cb[(size_t)row * Ndim + col] = (bf16_t)acc[sub][j][r];
        else
          Cf[(size_t)row * Ndim + col] = acc[sub][j][r] + bv;
      }
  }
}

// ---------------------------------------------------------------------------
// Flash attention (causal). Grid: (T/128, B*H), block 128 = 4 waves.
// Each wave owns 32 query rows (two 16-row fragments). Key steps of 32.
// V tiles are double-buffered in LDS via global_load_async_to_lds_b128 and
// drained with s_wait_asynccnt -> fetch overlaps QK^T + softmax.
// All LDS buffers are per-wave; DS ops are in-order per wave, so no barriers.
// ---------------------------------------------------------------------------
__global__ void __launch_bounds__(128)
attn_causal(const bf16_t* __restrict__ Q, const bf16_t* __restrict__ K,
            const bf16_t* __restrict__ V, bf16_t* __restrict__ O) {
  __shared__ __align__(16) bf16_t Vs[4][2][32][64];  // async staging, [s][d]
  __shared__ __align__(16) bf16_t Vt[4][64][40];     // transposed       [d][s]
  __shared__ __align__(16) bf16_t Pl[4][32][40];     // probabilities    [q][s]

  const int wave = threadIdx.x >> 5;
  const int lane = threadIdx.x & 31;
  const int half = lane >> 4;
  const int l16  = lane & 15;
  const int kb   = half * 8;

  const int bh = blockIdx.y;             // 0..B*H-1
  const int b  = bh >> 4;                // /H
  const int h  = bh & 15;
  const int qblock = blockIdx.x * 128;
  const int qbase  = qblock + wave * 32;

  const size_t base = (size_t)b * Tc * Cc + (size_t)h * Dc;
  const bf16_t* Qp = Q + base;
  const bf16_t* Kp = K + base;
  const bf16_t* Vp = V + base;
  bf16_t*       Op = O + base;

  // Q fragments: two 16-row halves, two 32-wide D chunks each.
  v16bf qf[2][2];
#pragma unroll
  for (int rf = 0; rf < 2; ++rf) {
    const bf16_t* qrow = Qp + (size_t)(qbase + rf * 16 + l16) * Cc;
    qf[rf][0] = load_frag(qrow + kb);
    qf[rf][1] = load_frag(qrow + 32 + kb);
  }

  float mst[2][8], lst[2][8];
  v8f acc[2][4] = {};
#pragma unroll
  for (int rf = 0; rf < 2; ++rf)
#pragma unroll
    for (int r = 0; r < 8; ++r) { mst[rf][r] = -1e30f; lst[rf][r] = 0.0f; }

  // Async V-tile prefetch: 32x64 bf16 = 256 x 16B chunks, 8 per lane.
  auto vprefetch = [&](int buf, int s0) {
#pragma unroll
    for (int i = 0; i < 8; ++i) {
      const int c = i * 32 + lane;
      const int s = c >> 3, d0 = (c & 7) * 8;
      unsigned ldsoff = (unsigned)(size_t)&Vs[wave][buf][s][d0];
      unsigned long long g =
          (unsigned long long)(size_t)&Vp[(size_t)(s0 + s) * Cc + d0];
      asm volatile("global_load_async_to_lds_b128 %0, %1, off"
                   :: "v"(ldsoff), "v"(g) : "memory");
    }
  };

  const int send = qblock + 128;   // uniform per block
  int cur = 0;
  vprefetch(cur, 0);

  for (int s0 = 0; s0 < send; s0 += 32) {
    const bool more = (s0 + 32) < send;
    if (more) vprefetch(cur ^ 1, s0 + 32);   // overlap with QK^T + softmax

    // ---- scores: rows [qbase..qbase+31] x cols [s0..s0+31]
    v8f sc[2][2];
#pragma unroll
    for (int j = 0; j < 2; ++j) {
      const bf16_t* krow = Kp + (size_t)(s0 + j * 16 + l16) * Cc;
      v16bf kf0 = load_frag(krow + kb);
      v16bf kf1 = load_frag(krow + 32 + kb);
#pragma unroll
      for (int rf = 0; rf < 2; ++rf) {
        v8f z = {};
        z = WMMA_BF16(qf[rf][0], kf0, z);
        sc[rf][j] = WMMA_BF16(qf[rf][1], kf1, z);
      }
    }

    // ---- scale + causal mask + online softmax
#pragma unroll
    for (int rf = 0; rf < 2; ++rf) {
      float alpha[8];
#pragma unroll
      for (int r = 0; r < 8; ++r) {
        const int row = qbase + rf * 16 + half * 8 + r;
#pragma unroll
        for (int j = 0; j < 2; ++j) {
          const int col = s0 + j * 16 + l16;
          float s = sc[rf][j][r] * 0.125f;   // 1/sqrt(64)
          sc[rf][j][r] = (col <= row) ? s : -1e30f;
        }
        float t = fmaxf(sc[rf][0][r], sc[rf][1][r]);
        t = fmaxf(t, __shfl_xor(t, 1));
        t = fmaxf(t, __shfl_xor(t, 2));
        t = fmaxf(t, __shfl_xor(t, 4));
        t = fmaxf(t, __shfl_xor(t, 8));
        const float mnew = fmaxf(mst[rf][r], t);
        alpha[r] = __expf(mst[rf][r] - mnew);
        const float p0 = __expf(sc[rf][0][r] - mnew);
        const float p1 = __expf(sc[rf][1][r] - mnew);
        sc[rf][0][r] = p0; sc[rf][1][r] = p1;
        float rs = p0 + p1;
        rs += __shfl_xor(rs, 1);
        rs += __shfl_xor(rs, 2);
        rs += __shfl_xor(rs, 4);
        rs += __shfl_xor(rs, 8);
        lst[rf][r] = lst[rf][r] * alpha[r] + rs;
        mst[rf][r] = mnew;
      }
#pragma unroll
      for (int j = 0; j < 4; ++j)
#pragma unroll
        for (int r = 0; r < 8; ++r) acc[rf][j][r] *= alpha[r];

      // P (C-layout) -> LDS so it can be reloaded as an A-fragment
#pragma unroll
      for (int j = 0; j < 2; ++j)
#pragma unroll
        for (int r = 0; r < 8; ++r)
          Pl[wave][rf * 16 + half * 8 + r][j * 16 + l16] = (bf16_t)sc[rf][j][r];
    }

    // ---- drain async V copy for this step (8 in flight from next step)
    if (more) asm volatile("s_wait_asynccnt 0x8" ::: "memory");
    else      asm volatile("s_wait_asynccnt 0x0" ::: "memory");

    // ---- transpose Vs[cur] (32s x 64d) -> Vt[d][s]
#pragma unroll
    for (int i = 0; i < 8; ++i) {
      const int c = i * 32 + lane;
      const int s = c >> 3, d0 = (c & 7) * 8;
      v8bf v = *(const v8bf*)&Vs[wave][cur][s][d0];
#pragma unroll
      for (int q = 0; q < 8; ++q) Vt[wave][d0 + q][s] = v[q];
    }
    asm volatile("" ::: "memory");   // keep DS program order (HW is in-order/wave)

    // ---- ctx += P(32x32) @ V(32x64)
    v16bf pf0 = load_frag(&Pl[wave][l16][kb]);
    v16bf pf1 = load_frag(&Pl[wave][16 + l16][kb]);
#pragma unroll
    for (int j = 0; j < 4; ++j) {
      v16bf vf = load_frag(&Vt[wave][j * 16 + l16][kb]);
      acc[0][j] = WMMA_BF16(pf0, vf, acc[0][j]);
      acc[1][j] = WMMA_BF16(pf1, vf, acc[1][j]);
    }
    cur ^= 1;
  }

  // ---- normalize and write ctx (bf16)
#pragma unroll
  for (int rf = 0; rf < 2; ++rf)
#pragma unroll
    for (int r = 0; r < 8; ++r) {
      const float inv = 1.0f / lst[rf][r];
      const int row = qbase + rf * 16 + half * 8 + r;
      bf16_t* op = Op + (size_t)row * Cc;
#pragma unroll
      for (int j = 0; j < 4; ++j)
        op[j * 16 + l16] = (bf16_t)(acc[rf][j][r] * inv);
    }
}

// ---------------------------------------------------------------------------
// Launch: x,Wq,Wk,Wv,Wo,bo (f32 in) -> out f32 [B,T,C]
// ---------------------------------------------------------------------------
extern "C" void kernel_launch(void* const* d_in, const int* in_sizes, int n_in,
                              void* d_out, int out_size, void* d_ws, size_t ws_size,
                              hipStream_t stream) {
  const float* x  = (const float*)d_in[0];
  const float* Wq = (const float*)d_in[1];
  const float* Wk = (const float*)d_in[2];
  const float* Wv = (const float*)d_in[3];
  const float* Wo = (const float*)d_in[4];
  const float* bo = (const float*)d_in[5];
  float* out = (float*)d_out;

  char* ws = (char*)d_ws;
  size_t off = 0;
  const size_t szX = (size_t)Mrows * Cc * sizeof(bf16_t);
  const size_t szW = (size_t)Cc * Cc * sizeof(bf16_t);
  bf16_t* xb  = (bf16_t*)(ws + off); off += szX;
  bf16_t* Wqt = (bf16_t*)(ws + off); off += szW;   // transposed bf16 weights
  bf16_t* Wkt = (bf16_t*)(ws + off); off += szW;
  bf16_t* Wvt = (bf16_t*)(ws + off); off += szW;
  bf16_t* Wot = (bf16_t*)(ws + off); off += szW;
  bf16_t* Qb  = (bf16_t*)(ws + off); off += szX;
  bf16_t* Kb  = (bf16_t*)(ws + off); off += szX;
  bf16_t* Vb  = (bf16_t*)(ws + off); off += szX;
  bf16_t* Ctx = (bf16_t*)(ws + off); off += szX;

  // 1) conversions: x linear, weights transposed ([k][n] -> [n][k])
  {
    int nX = Mrows * Cc;
    cvt_f32_bf16<<<(nX + 255) / 256, 256, 0, stream>>>(x, xb, nX);
    dim3 tg(Cc / 32, Cc / 32);
    cvt_transpose_f32_bf16<<<tg, 256, 0, stream>>>(Wq, Wqt, Cc, Cc);
    cvt_transpose_f32_bf16<<<tg, 256, 0, stream>>>(Wk, Wkt, Cc, Cc);
    cvt_transpose_f32_bf16<<<tg, 256, 0, stream>>>(Wv, Wvt, Cc, Cc);
    cvt_transpose_f32_bf16<<<tg, 256, 0, stream>>>(Wo, Wot, Cc, Cc);
  }

  // 2) Q/K/V projections (bf16 out)
  dim3 ggrid(Cc / 128, Mrows / 128);
  gemm_abt<true><<<ggrid, 256, 0, stream>>>(xb, Wqt, Qb, nullptr, nullptr, Mrows, Cc, Cc);
  gemm_abt<true><<<ggrid, 256, 0, stream>>>(xb, Wkt, Kb, nullptr, nullptr, Mrows, Cc, Cc);
  gemm_abt<true><<<ggrid, 256, 0, stream>>>(xb, Wvt, Vb, nullptr, nullptr, Mrows, Cc, Cc);

  // 3) causal flash attention -> Ctx (bf16, [B*T, C])
  attn_causal<<<dim3(Tc / 128, Bc * Hc), 128, 0, stream>>>(Qb, Kb, Vb, Ctx);

  // 4) output projection + bias (f32 out)
  gemm_abt<false><<<ggrid, 256, 0, stream>>>(Ctx, Wot, nullptr, out, bo, Mrows, Cc, Cc);
}
